// MiniMaxLightningAttention_27101243638161
// MI455X (gfx1250) — matrix operand order repaired and checked by reading, not compiled, not run on hardware
//
#include <hip/hip_runtime.h>
#include <hip/hip_bf16.h>
#include <stdint.h>

typedef unsigned short u16;
typedef unsigned int u32;
typedef __bf16 bf16_t;
typedef bf16_t v16bf __attribute__((ext_vector_type(16)));
typedef float v8f __attribute__((ext_vector_type(8)));

#define NUM_TOKENS 8192
#define HIDDEN 4096
#define N_HEADS 32
#define N_KV_HEADS 8
#define HEAD_DIM 128
#define QKV_DIM 6144           // (32 + 2*8) * 128
#define Q_SIZE 4096
#define KV_SIZE 1024
#define ROT_HALF 32            // ROTARY_DIM/2
#define BK 64                  // K tile (two 32-wide WMMA k-steps)

// ---------------- helpers ----------------
__device__ __forceinline__ u16 f2bf(float f) {
    u32 u = __float_as_uint(f);
    u32 r = u + 0x7FFFu + ((u >> 16) & 1u);   // round-to-nearest-even
    return (u16)(r >> 16);
}
__device__ __forceinline__ float elu1(float x) {
    return x > 0.0f ? x + 1.0f : expf(x);     // elu(x)+1
}

// async global -> LDS copy, 16B per lane, ASYNCcnt-tracked (CDNA5)
__device__ __forceinline__ void async_ld16(u32 ldsAddr, const u16* sbase, u32 byteOff) {
    asm volatile("global_load_async_to_lds_b128 %0, %1, %2 offset:0"
                 :: "v"(ldsAddr), "v"(byteOff), "s"(sbase)
                 : "memory");
}
__device__ __forceinline__ void wait_async0() {
    asm volatile("s_wait_asynccnt 0x0" ::: "memory");
}

// ---------------- f32 -> bf16 bulk convert (8 elems/thread, exact grids) ----
__global__ __launch_bounds__(256) void cvt_bf16_kernel(const float* __restrict__ in,
                                                       u16* __restrict__ out) {
    size_t i = (size_t)blockIdx.x * blockDim.x + threadIdx.x;
    const float4* p = (const float4*)in + i * 2;
    float4 a = p[0], b = p[1];
    union { u16 s[8]; uint4 q; } u;
    u.s[0] = f2bf(a.x); u.s[1] = f2bf(a.y); u.s[2] = f2bf(a.z); u.s[3] = f2bf(a.w);
    u.s[4] = f2bf(b.x); u.s[5] = f2bf(b.y); u.s[6] = f2bf(b.z); u.s[7] = f2bf(b.w);
    ((uint4*)out)[i] = u.q;
}

// ---------------- bf16 WMMA GEMM: C[M,N] = A[M,K] * B[N,K]^T ----------------
// Block tile 128x128x64, 256 threads = 8 waves (2 M x 4 N), wave tile 64x32.
// Double-buffered LDS filled with global_load_async_to_lds_b128.
__global__ __launch_bounds__(256)
void gemm_bf16_kernel(const u16* __restrict__ A, const u16* __restrict__ B,
                      float* __restrict__ C, int M, int N, int K) {
    __shared__ __align__(16) u16 As[2][128 * BK];   // 2 x 16 KiB
    __shared__ __align__(16) u16 Bs[2][128 * BK];   // 2 x 16 KiB

    const int tid  = threadIdx.x;
    const int lane = tid & 31;            // wave32
    const int wave = tid >> 5;
    const int wm   = (wave & 1) * 64;     // wave M offset
    const int wn   = (wave >> 1) * 32;    // wave N offset
    const int rowBase = blockIdx.y * 128;
    const int colBase = blockIdx.x * 128;
    const bool hi  = lane >= 16;
    const int l16  = lane & 15;

    v8f acc[4][2];
    const v8f vzero = {0.f,0.f,0.f,0.f,0.f,0.f,0.f,0.f};
    #pragma unroll
    for (int mt = 0; mt < 4; ++mt)
        #pragma unroll
        for (int nt = 0; nt < 2; ++nt) acc[mt][nt] = vzero;

    // staging: thread t owns row (t/2), 64B half (t&1)*32 elems -> 4 x b128
    const int srow = tid >> 1;
    const int scol = (tid & 1) * 32;
    const u32 aOffBase = ((u32)(rowBase + srow) * (u32)K + (u32)scol) * 2u;
    const u32 bOffBase = ((u32)(colBase + srow) * (u32)K + (u32)scol) * 2u;
    const u32 ldsStage = (u32)(srow * BK + scol) * 2u;       // bytes within buf
    const u32 aLds0 = (u32)(uintptr_t)&As[0][0] + ldsStage;  // flat[31:0] == LDS offset
    const u32 bLds0 = (u32)(uintptr_t)&Bs[0][0] + ldsStage;
    const u32 bufStride = 128u * BK * 2u;                    // 16384 B

    const int nk = K / BK;

    // prologue: fill buffer 0
    {
        #pragma unroll
        for (int c = 0; c < 4; ++c) {
            async_ld16(aLds0 + c * 16u, A, aOffBase + c * 16u);
            async_ld16(bLds0 + c * 16u, B, bOffBase + c * 16u);
        }
        wait_async0();
        __syncthreads();
    }

    for (int t = 0; t < nk; ++t) {
        const int cur = t & 1;
        // issue async loads for next tile into the idle buffer (overlaps WMMAs)
        if (t + 1 < nk) {
            const u32 go = (u32)(t + 1) * (BK * 2u);
            const u32 la = aLds0 + (u32)(cur ^ 1) * bufStride;
            const u32 lb = bLds0 + (u32)(cur ^ 1) * bufStride;
            #pragma unroll
            for (int c = 0; c < 4; ++c) {
                async_ld16(la + c * 16u, A, aOffBase + go + c * 16u);
                async_ld16(lb + c * 16u, B, bOffBase + go + c * 16u);
            }
        }
        // compute from current buffer: 2 k-substeps x (4 M x 2 N) WMMAs
        const u16* asb = &As[cur][0];
        const u16* bsb = &Bs[cur][0];
        #pragma unroll
        for (int ks = 0; ks < BK; ks += 32) {
            v16bf bfrag[2];
            #pragma unroll
            for (int nt = 0; nt < 2; ++nt) {
                const u16* bp = bsb + (wn + nt * 16 + l16) * BK + ks + (hi ? 16 : 0);
                union { v16bf v; uint4 q[2]; } u;
                u.q[0] = *(const uint4*)bp;
                u.q[1] = *(const uint4*)(bp + 8);
                bfrag[nt] = u.v;
            }
            #pragma unroll
            for (int mt = 0; mt < 4; ++mt) {
                const u16* ap = asb + (wm + mt * 16 + l16) * BK + ks + (hi ? 16 : 0);
                union { v16bf v; uint4 q[2]; } u;
                u.q[0] = *(const uint4*)ap;
                u.q[1] = *(const uint4*)(ap + 8);
                v16bf afrag = u.v;
                #pragma unroll
                for (int nt = 0; nt < 2; ++nt)
                    acc[mt][nt] = __builtin_amdgcn_wmma_f32_16x16x32_bf16(
                        false, afrag, false, bfrag[nt], (short)0, acc[mt][nt],
                        false, false);
            }
        }
        if (t + 1 < nk) {
            wait_async0();        // this wave's next-tile loads landed in LDS
            __syncthreads();      // all waves done reading cur + loading next
        }
    }

    // C layout: lanes0-15 N=lane, M=vgpr r; lanes16-31 N=lane-16, M=8+r
    const int hi8 = hi ? 8 : 0;
    #pragma unroll
    for (int mt = 0; mt < 4; ++mt)
        #pragma unroll
        for (int nt = 0; nt < 2; ++nt) {
            int col = colBase + wn + nt * 16 + l16;
            float* cp = C + (size_t)(rowBase + wm + mt * 16 + hi8) * N + col;
            #pragma unroll
            for (int r = 0; r < 8; ++r) cp[(size_t)r * N] = acc[mt][nt][r];
        }
}

// ---------------- RoPE (partial, rotary_dim=64) + elu+1 on q,k (in place) ---
__global__ __launch_bounds__(256)
void rope_elu_kernel(float* __restrict__ qkv, const int* __restrict__ positions) {
    const int b = blockIdx.x;
    __shared__ float cs[ROT_HALF], sn[ROT_HALF];
    const int tid = threadIdx.x;
    if (tid < ROT_HALF) {
        float p = (float)positions[b];
        // inv_freq = 1e7 ^ (-i/32)
        float inv = expf(-(float)tid * (1.0f / 32.0f) * 16.11809565095832f); // ln(1e7)
        float ang = p * inv;
        cs[tid] = cosf(ang);
        sn[tid] = sinf(ang);
    }
    __syncthreads();
    float* base = qkv + (size_t)b * QKV_DIM;
    // 40 heads (32 q + 8 k), rotary pairs (i, i+32), i in [0,32)
    for (int idx = tid; idx < 40 * 32; idx += 256) {
        int h = idx >> 5, i = idx & 31;
        float* hp = base + (h < 32 ? h * 128 : Q_SIZE + (h - 32) * 128);
        float x1 = hp[i], x2 = hp[i + 32];
        float c = cs[i], s = sn[i];
        hp[i]      = elu1(x1 * c - x2 * s);
        hp[i + 32] = elu1(x2 * c + x1 * s);
    }
    // pass-through dims 64..127: elu+1 only
    for (int idx = tid; idx < 40 * 64; idx += 256) {
        int h = idx >> 6, i = 64 + (idx & 63);
        float* hp = base + (h < 32 ? h * 128 : Q_SIZE + (h - 32) * 128);
        hp[i] = elu1(hp[i]);
    }
}

__global__ void zero_kernel(float* p, int n) {
    int i = blockIdx.x * blockDim.x + threadIdx.x;
    if (i < n) p[i] = 0.0f;
}

// ---------------- k_sum[j,d] = sum_b k[b,j,d] (post rope+elu) ---------------
__global__ __launch_bounds__(128)
void ksum_kernel(const float* __restrict__ qkv, float* __restrict__ ksum) {
    const int j = blockIdx.y;                 // kv head
    const int d = threadIdx.x;                // dim
    const float* p = qkv + (size_t)blockIdx.x * 256 * QKV_DIM + Q_SIZE + j * 128 + d;
    float s = 0.0f;
    for (int t = 0; t < 256; ++t) s += p[(size_t)t * QKV_DIM];
    atomicAdd(&ksum[j * 128 + d], s);
}

// ---------------- per-token rank-8 attention + normalize -> bf16 attn ------
__global__ __launch_bounds__(256)
void attn_kernel(const float* __restrict__ qkv, const float* __restrict__ ksum,
                 u16* __restrict__ attn) {
    const int b = blockIdx.x;
    __shared__ float Q[32 * 128];
    __shared__ float Ks[8 * 128];
    __shared__ float Vs[8 * 128];
    __shared__ float S[32 * 8];
    __shared__ float nrm[32];
    const int tid = threadIdx.x;
    const float* base = qkv + (size_t)b * QKV_DIM;
    for (int i = tid; i < 4096; i += 256) Q[i]  = base[i];
    for (int i = tid; i < 1024; i += 256) Ks[i] = base[Q_SIZE + i];
    for (int i = tid; i < 1024; i += 256) Vs[i] = base[Q_SIZE + KV_SIZE + i];
    __syncthreads();
    {   // S[n,j] = q_n . k_j   (256 threads <-> 32x8)
        int n = tid >> 3, j = tid & 7;
        float s = 0.0f;
        #pragma unroll 8
        for (int d = 0; d < 128; ++d) s += Q[n * 128 + d] * Ks[j * 128 + d];
        S[n * 8 + j] = s;
    }
    if (tid < 32) {  // normalizer[n] = q_n . ksum[n/4] + 1e-6
        const float* kp = ksum + (tid >> 2) * 128;
        float s = 0.0f;
        #pragma unroll 8
        for (int d = 0; d < 128; ++d) s += Q[tid * 128 + d] * kp[d];
        nrm[tid] = s + 1e-6f;
    }
    __syncthreads();
    // out[n,v] = 4 * sum_j S[n,j] * V[j,v]; attn = out / nrm
    const int n = tid >> 3, v0 = (tid & 7) * 16;
    const float inv = 4.0f / nrm[n];
    float sc[8];
    #pragma unroll
    for (int j = 0; j < 8; ++j) sc[j] = S[n * 8 + j];
    u16* op = attn + (size_t)b * 4096 + n * 128 + v0;
    #pragma unroll
    for (int i = 0; i < 16; ++i) {
        float o = 0.0f;
        #pragma unroll
        for (int j = 0; j < 8; ++j) o += sc[j] * Vs[j * 128 + v0 + i];
        op[i] = f2bf(o * inv);
    }
}

// ---------------- launcher ----------------
extern "C" void kernel_launch(void* const* d_in, const int* in_sizes, int n_in,
                              void* d_out, int out_size, void* d_ws, size_t ws_size,
                              hipStream_t stream) {
    const float* hidden    = (const float*)d_in[0];   // [8192,4096]
    const int*   positions = (const int*)d_in[1];     // [8192]
    const float* w_qkv     = (const float*)d_in[2];   // [6144,4096]
    const float* w_o       = (const float*)d_in[3];   // [4096,4096]
    float* out = (float*)d_out;                       // [8192,4096]

    char* ws = (char*)d_ws;
    size_t o = 0;
    u16*   Xbf    = (u16*)(ws + o);  o += (size_t)NUM_TOKENS * HIDDEN * 2;   // 64 MiB
    u16*   Wqkvbf = (u16*)(ws + o);  o += (size_t)QKV_DIM * HIDDEN * 2;      // 48 MiB
    u16*   Wobf   = (u16*)(ws + o);  o += (size_t)HIDDEN * HIDDEN * 2;       // 32 MiB
    float* qkv    = (float*)(ws + o); o += (size_t)NUM_TOKENS * QKV_DIM * 4; // 192 MiB
    u16*   attn   = (u16*)(ws + o);  o += (size_t)NUM_TOKENS * HIDDEN * 2;   // 64 MiB
    float* ksum   = (float*)(ws + o); o += 1024 * 4;

    // 1) convert inputs to bf16 (exact grids: n/8/256)
    cvt_bf16_kernel<<<16384, 256, 0, stream>>>(hidden, Xbf);
    cvt_bf16_kernel<<<12288, 256, 0, stream>>>(w_qkv, Wqkvbf);
    cvt_bf16_kernel<<< 8192, 256, 0, stream>>>(w_o, Wobf);

    // 2) QKV GEMM: [8192,6144] = X * Wqkv^T
    gemm_bf16_kernel<<<dim3(QKV_DIM / 128, NUM_TOKENS / 128), 256, 0, stream>>>(
        Xbf, Wqkvbf, qkv, NUM_TOKENS, QKV_DIM, HIDDEN);

    // 3) RoPE + elu+1 on q,k (in place)
    rope_elu_kernel<<<NUM_TOKENS, 256, 0, stream>>>(qkv, positions);

    // 4) k_sum reduction
    zero_kernel<<<4, 256, 0, stream>>>(ksum, 1024);
    ksum_kernel<<<dim3(NUM_TOKENS / 256, N_KV_HEADS), 128, 0, stream>>>(qkv, ksum);

    // 5) per-token rank-8 attention -> bf16 attn
    attn_kernel<<<NUM_TOKENS, 256, 0, stream>>>(qkv, ksum, attn);

    // 6) output GEMM: [8192,4096] = attn * Wo^T
    gemm_bf16_kernel<<<dim3(HIDDEN / 128, NUM_TOKENS / 128), 256, 0, stream>>>(
        attn, Wobf, out, NUM_TOKENS, HIDDEN, HIDDEN);
}
